// ChunkedAttention_53094385713647
// MI455X (gfx1250) — matrix operand order repaired and checked by reading, not compiled, not run on hardware
//
#include <hip/hip_runtime.h>
#include <hip/hip_bf16.h>

// ---------------------------------------------------------------------------
// MI455X (gfx1250) multi-head attention, bf16 WMMA path.
//   x[2,2048,1024] f32 -> Q/K/V proj (WMMA bf16) -> streaming attention
//   (WMMA bf16, ones-matrix rowsum trick) -> output proj (WMMA bf16, f32 out)
// Round 2: fatter wave tiles. GEMM wave = 64x64 (16 WMMA / 16 b128 loads per
// k-step). Attention wave = 32 q-rows sharing K/V fragments (18 WMMA / 16 global + 4 LDS b128
// loads per 32 KV columns).
// ---------------------------------------------------------------------------

typedef __bf16 bf16_t;
typedef __attribute__((ext_vector_type(16))) __bf16 v16bf;
typedef __attribute__((ext_vector_type(8)))  __bf16 v8bf;
typedef __attribute__((ext_vector_type(8)))  float  v8f;

#define DMODEL 1024
#define SEQ    2048
#define NHEAD  16
#define DK     64
#define MROWS  4096   // B*S

static __device__ inline v8f wmma_bf16(v16bf a, v16bf b, v8f c) {
    // (neg_a, A, neg_b, B, c_mod, C, reuse_a, reuse_b)
    return __builtin_amdgcn_wmma_f32_16x16x32_bf16(false, a, false, b,
                                                   (short)0, c, false, false);
}

static __device__ inline v16bf make16(v8bf lo, v8bf hi) {
    v16bf r;
#pragma unroll
    for (int i = 0; i < 8; ++i) { r[i] = lo[i]; r[i + 8] = hi[i]; }
    return r;
}

// A-matrix fragment (16x32 bf16, M x K). Per ISA layout:
// lane group g = lane>>4, row m = lane&15 (caller bakes m into rowptr).
// element e: K = kbase + (e<8 ? 8*g + e : 16 + 8*g + (e-8))
static __device__ inline v16bf load_a_frag(const bf16_t* rowptr, int g, int kbase) {
    v8bf lo = *(const v8bf*)(rowptr + kbase + 8 * g);
    v8bf hi = *(const v8bf*)(rowptr + kbase + 16 + 8 * g);
    return make16(lo, hi);
}

// B-matrix fragment (32x16 bf16, K x N). Per ISA layout: lane = column n,
// element e: K = kbase + (e<8 ? 8*g + e : 16 + 8*g + (e-8)) (same as A).
// Caller bakes column n into rowptr (memory row holding that column's K-run).
static __device__ inline v16bf load_b_frag(const bf16_t* rowptr, int g, int kbase) {
    v8bf lo = *(const v8bf*)(rowptr + kbase + 8 * g);
    v8bf hi = *(const v8bf*)(rowptr + kbase + 16 + 8 * g);
    return make16(lo, hi);
}

// ---------------------------------------------------------------------------
// f32 -> bf16 cast
// ---------------------------------------------------------------------------
__global__ void cast_f32_bf16(const float* __restrict__ src,
                              bf16_t* __restrict__ dst, int n) {
    int i = blockIdx.x * blockDim.x + threadIdx.x;
    if (i < n) dst[i] = (bf16_t)src[i];
}

// ---------------------------------------------------------------------------
// C[M=4096, N=1024] = A_bf16 @ W_bf16^T + bias, epilogue modes:
//   mode 0: f32 out, row-major [M,N]              (final projection)
//   mode 1: bf16 out, [B,H,S,Dk]  * scale         (Q with scale=1/8, K)
//   mode 2: bf16 out, [B,H,Dk,S]  (transposed V)
// Block: 256 thr = 8 waves (2x4) -> 128x256 tile; wave -> 64x64 (4x4 WMMA).
// ---------------------------------------------------------------------------
static __device__ inline void store_tile(v8f acc, int mbase, int nbase,
                                         int g, int ln,
                                         const float* __restrict__ bias,
                                         void* __restrict__ out,
                                         int mode, float scale) {
    const int n = nbase + ln;
    const float bv = bias[n];
#pragma unroll
    for (int r = 0; r < 8; ++r) {
        const int m = mbase + r + 8 * g;
        const float v = (acc[r] + bv) * scale;
        if (mode == 0) {
            ((float*)out)[(size_t)m * DMODEL + n] = v;
        } else {
            const int b = m >> 11, s = m & (SEQ - 1);
            const int h = n >> 6, d = n & (DK - 1);
            size_t idx;
            if (mode == 1) idx = ((size_t)(b * NHEAD + h) * SEQ + s) * DK + d;
            else           idx = ((size_t)(b * NHEAD + h) * DK + d) * SEQ + s;
            ((bf16_t*)out)[idx] = (bf16_t)v;
        }
    }
}

__global__ __launch_bounds__(256) void gemm_bf16(
    const bf16_t* __restrict__ A, const bf16_t* __restrict__ W,
    const float* __restrict__ bias, void* __restrict__ out,
    int mode, float scale) {
    const int lane = threadIdx.x & 31;
    const int wave = threadIdx.x >> 5;
    const int g  = lane >> 4;
    const int ln = lane & 15;

    const int m0 = blockIdx.x * 128 + (wave & 1) * 64;
    const int n0 = blockIdx.y * 256 + (wave >> 1) * 64;

    const bf16_t* arow = A + (size_t)(m0 + ln) * DMODEL;
    const bf16_t* brow = W + (size_t)(n0 + ln) * DMODEL;

    v8f acc[4][4];
#pragma unroll
    for (int i = 0; i < 4; ++i)
#pragma unroll
        for (int j = 0; j < 4; ++j) acc[i][j] = (v8f){};

#pragma unroll 2
    for (int k = 0; k < DMODEL; k += 32) {
        v16bf a[4], b[4];
#pragma unroll
        for (int i = 0; i < 4; ++i)
            a[i] = load_a_frag(arow + (size_t)(16 * i) * DMODEL, g, k);
#pragma unroll
        for (int j = 0; j < 4; ++j)
            b[j] = load_b_frag(brow + (size_t)(16 * j) * DMODEL, g, k);
#pragma unroll
        for (int i = 0; i < 4; ++i)
#pragma unroll
            for (int j = 0; j < 4; ++j)
                acc[i][j] = wmma_bf16(a[i], b[j], acc[i][j]);
    }

#pragma unroll
    for (int i = 0; i < 4; ++i)
#pragma unroll
        for (int j = 0; j < 4; ++j)
            store_tile(acc[i][j], m0 + 16 * i, n0 + 16 * j, g, ln,
                       bias, out, mode, scale);
}

// ---------------------------------------------------------------------------
// Streaming attention. Q already scaled by 1/sqrt(Dk) and biased.
// Q,K: bf16 [B,H,S,Dk]; Vt: bf16 [B,H,Dk,S]; Aout: bf16 [B,S,H*Dk].
// Block = 128 thr (4 waves); wave owns 32 q rows (2 tiles sharing K/V frags);
// block owns 128 q rows. Per 32 KV columns & wave:
//   8 WMMA (scores) + exp + LDS relayout + 2 WMMA (rowsum via ones matrix)
//   + 8 WMMA (P.V). Final: elementwise divide by rowsum.
// ---------------------------------------------------------------------------
__global__ __launch_bounds__(128) void attn_kernel(
    const bf16_t* __restrict__ Q, const bf16_t* __restrict__ K,
    const bf16_t* __restrict__ Vt, bf16_t* __restrict__ Aout) {
    __shared__ bf16_t ptile[4][2][16 * 32];  // per-wave, per-qtile P, stride 32

    const int lane = threadIdx.x & 31;
    const int wave = threadIdx.x >> 5;
    const int g  = lane >> 4;
    const int ln = lane & 15;

    const int bh    = blockIdx.x >> 4;                   // b*NHEAD + h
    const int qbase = (blockIdx.x & 15) * 128 + wave * 32;

    v16bf qa[2][2];
#pragma unroll
    for (int u = 0; u < 2; ++u) {
        const bf16_t* qrow = Q + ((size_t)bh * SEQ + qbase + 16 * u + ln) * DK;
        qa[u][0] = load_a_frag(qrow, g, 0);
        qa[u][1] = load_a_frag(qrow, g, 32);
    }

    v16bf vones;
#pragma unroll
    for (int i = 0; i < 16; ++i) vones[i] = (bf16_t)1.0f;

    v8f acc[2][4], den[2];
#pragma unroll
    for (int u = 0; u < 2; ++u) {
        den[u] = (v8f){};
#pragma unroll
        for (int d = 0; d < 4; ++d) acc[u][d] = (v8f){};
    }

    const bf16_t* krow  = K  + ((size_t)bh * SEQ + ln) * DK;
    const bf16_t* vtrow = Vt + ((size_t)bh * DK + ln) * SEQ;

    for (int kk = 0; kk < SEQ; kk += 32) {
#pragma unroll
        for (int t = 0; t < 2; ++t) {
            const bf16_t* kr = krow + (size_t)(kk + 16 * t) * DK;
            v16bf b0 = load_b_frag(kr, g, 0);
            v16bf b1 = load_b_frag(kr, g, 32);
#pragma unroll
            for (int u = 0; u < 2; ++u) {
                v8f s = {};
                s = wmma_bf16(qa[u][0], b0, s);
                s = wmma_bf16(qa[u][1], b1, s);
                bf16_t* pt = &ptile[wave][u][0];
#pragma unroll
                for (int r = 0; r < 8; ++r)
                    pt[(r + 8 * g) * 32 + 16 * t + ln] = (bf16_t)__expf(s[r]);
            }
        }
        // wave-private LDS tiles: order stores before the transposed reload
        asm volatile("s_wait_dscnt 0" ::: "memory");

        v16bf pa[2];
#pragma unroll
        for (int u = 0; u < 2; ++u) {
            pa[u] = load_a_frag(&ptile[wave][u][0] + ln * 32, g, 0);
            den[u] = wmma_bf16(pa[u], vones, den[u]);        // row sums
        }
#pragma unroll
        for (int dt = 0; dt < 4; ++dt) {
            v16bf vb = load_b_frag(vtrow + (size_t)(16 * dt) * SEQ, g, kk);
#pragma unroll
            for (int u = 0; u < 2; ++u)
                acc[u][dt] = wmma_bf16(pa[u], vb, acc[u][dt]);
        }
    }

    const int b = bh >> 4, h = bh & (NHEAD - 1);
#pragma unroll
    for (int u = 0; u < 2; ++u) {
#pragma unroll
        for (int r = 0; r < 8; ++r) {
            const int srow = qbase + 16 * u + r + 8 * g;
            const float inv = 1.0f / den[u][r];
            bf16_t* o = Aout + ((size_t)(b * SEQ + srow)) * DMODEL + h * DK + ln;
            o[0]  = (bf16_t)(acc[u][0][r] * inv);
            o[16] = (bf16_t)(acc[u][1][r] * inv);
            o[32] = (bf16_t)(acc[u][2][r] * inv);
            o[48] = (bf16_t)(acc[u][3][r] * inv);
        }
    }
}

// ---------------------------------------------------------------------------
// Host side
// ---------------------------------------------------------------------------
extern "C" void kernel_launch(void* const* d_in, const int* in_sizes, int n_in,
                              void* d_out, int out_size, void* d_ws, size_t ws_size,
                              hipStream_t stream) {
    (void)in_sizes; (void)n_in; (void)out_size; (void)ws_size;

    const float* x  = (const float*)d_in[0];
    const float* wq = (const float*)d_in[1];
    const float* bq = (const float*)d_in[2];
    const float* wk = (const float*)d_in[3];
    const float* bk = (const float*)d_in[4];
    const float* wv = (const float*)d_in[5];
    const float* bv = (const float*)d_in[6];
    const float* wo = (const float*)d_in[7];
    const float* bo = (const float*)d_in[8];

    char* ws = (char*)d_ws;
    const size_t SZ_X = (size_t)MROWS * DMODEL * sizeof(bf16_t);   // 8 MB
    const size_t SZ_W = (size_t)DMODEL * DMODEL * sizeof(bf16_t);  // 2 MB
    bf16_t* xb  = (bf16_t*)(ws);                    // 8 MB
    bf16_t* wqb = (bf16_t*)(ws + SZ_X);             // 2 MB
    bf16_t* wkb = (bf16_t*)(ws + SZ_X + SZ_W);
    bf16_t* wvb = (bf16_t*)(ws + SZ_X + 2 * SZ_W);
    bf16_t* wob = (bf16_t*)(ws + SZ_X + 3 * SZ_W);
    bf16_t* Qb  = (bf16_t*)(ws + SZ_X + 4 * SZ_W);
    bf16_t* Kb  = (bf16_t*)(ws + 2 * SZ_X + 4 * SZ_W);
    bf16_t* Vtb = (bf16_t*)(ws + 3 * SZ_X + 4 * SZ_W);
    bf16_t* aob = (bf16_t*)(ws + 4 * SZ_X + 4 * SZ_W);

    const int NX = MROWS * DMODEL;       // 4,194,304
    const int NW = DMODEL * DMODEL;      // 1,048,576
    cast_f32_bf16<<<(NX + 255) / 256, 256, 0, stream>>>(x,  xb,  NX);
    cast_f32_bf16<<<(NW + 255) / 256, 256, 0, stream>>>(wq, wqb, NW);
    cast_f32_bf16<<<(NW + 255) / 256, 256, 0, stream>>>(wk, wkb, NW);
    cast_f32_bf16<<<(NW + 255) / 256, 256, 0, stream>>>(wv, wvb, NW);
    cast_f32_bf16<<<(NW + 255) / 256, 256, 0, stream>>>(wo, wob, NW);

    dim3 ggrid(MROWS / 128, DMODEL / 256);
    gemm_bf16<<<ggrid, 256, 0, stream>>>(xb, wqb, bq, Qb,  1, 0.125f); // Q/8
    gemm_bf16<<<ggrid, 256, 0, stream>>>(xb, wkb, bk, Kb,  1, 1.0f);
    gemm_bf16<<<ggrid, 256, 0, stream>>>(xb, wvb, bv, Vtb, 2, 1.0f);  // V^T

    attn_kernel<<<2 * NHEAD * (SEQ / 128), 128, 0, stream>>>(Qb, Kb, Vtb, aob);

    gemm_bf16<<<ggrid, 256, 0, stream>>>(aob, wob, bo, d_out, 0, 1.0f);
}
